// SelfAttentionLayer_40261023433059
// MI455X (gfx1250) — compile-verified, matrix operand order checked by reading, not run on hardware
//
#include <hip/hip_runtime.h>
#include <hip/hip_bf16.h>

typedef _Float16 h16;
typedef __attribute__((ext_vector_type(16))) _Float16 v16h;
typedef __attribute__((ext_vector_type(8)))  float    v8f;
typedef unsigned int u32x4 __attribute__((ext_vector_type(4)));
typedef int          i32x4 __attribute__((ext_vector_type(4)));
typedef int          i32x8 __attribute__((ext_vector_type(8)));

#define BATCH 4
#define NSEQ  4096        // 64*64 tokens per image
#define CDIM  64
#define DHEAD 8
#define DPAD  32          // pad head dim 8 -> 32 (zeros): one 16x16x32 WMMA covers QK^T
#define BM    64          // query rows per workgroup (4 waves x 16)
#define BK    64          // keys per chunk (4 QK + 8 PV WMMAs per iteration)
#define NWAVE 4

#if defined(__has_builtin)
#  if __has_builtin(__builtin_amdgcn_tensor_load_to_lds) && __has_builtin(__builtin_amdgcn_s_wait_tensorcnt)
#    define HAVE_TDM 1
#  endif
#endif
#ifndef HAVE_TDM
#  define HAVE_TDM 0
#endif

// ---------------- lane-shuffle reductions (wave32, ds_swizzle XOR) -------------
template<int MASK>
__device__ __forceinline__ float swz(float x) {
    return __int_as_float(__builtin_amdgcn_ds_swizzle(__float_as_int(x), (MASK << 10) | 0x1f));
}
__device__ __forceinline__ float rmax16(float v) {
    v = fmaxf(v, swz<1>(v)); v = fmaxf(v, swz<2>(v));
    v = fmaxf(v, swz<4>(v)); v = fmaxf(v, swz<8>(v));
    return v;
}
__device__ __forceinline__ float rsum16(float v) {
    v += swz<1>(v); v += swz<2>(v); v += swz<4>(v); v += swz<8>(v);
    return v;
}

#if HAVE_TDM
// ---- Tensor Data Mover: 2D tile (16-bit elements) global -> LDS ---------------
// D# layout per CDNA5 ISA ch.8: group0 = flags/lds_addr/global_addr/type,
// group1 = data_size, tensor dims, tile dims, dim0 stride. Groups 2/3 unused (2D).
// This toolchain's builtin takes 6 args: (g0, g1, g2, g3, g4, cpol).
__device__ __forceinline__ void tdm_load_2d(unsigned lds_off, const void* gaddr,
                                            unsigned tile_d0, unsigned tile_d1,
                                            unsigned tensor_d0, unsigned tensor_d1,
                                            unsigned stride0_elems) {
    const unsigned long long ga = (unsigned long long)(uintptr_t)gaddr;
    u32x4 g0;
    g0.x = 1u;                                                // count=1 (valid user D#)
    g0.y = lds_off;                                           // lds_addr (bytes)
    g0.z = (unsigned)(ga & 0xffffffffu);                      // global_addr[31:0]
    g0.w = (unsigned)((ga >> 32) & 0x01ffffffu) | (2u << 30); // global_addr[56:32] | type=2
    i32x8 g1;
    g1[0] = (int)(1u << 16);                                  // wg_mask=0, data_size=1 (2B)
    g1[1] = (int)((tensor_d0 & 0xffffu) << 16);               // tensor_dim0[15:0]
    g1[2] = (int)((tensor_d0 >> 16) | ((tensor_d1 & 0xffffu) << 16));
    g1[3] = (int)((tensor_d1 >> 16) | (tile_d0 << 16));       // tile_dim0
    g1[4] = (int)(tile_d1 & 0xffffu);                         // tile_dim1 (tile_dim2=0)
    g1[5] = (int)stride0_elems;                               // tensor_dim0_stride[31:0]
    g1[6] = 0;
    g1[7] = 0;
    const i32x4 z4 = {};
    const i32x8 z8 = {};
    __builtin_amdgcn_tensor_load_to_lds(g0, g1, z4, z4, z8, 0);
}
#endif

// ---------------- kernel 1: Q/K/V projections (1x1 convs), f32 -> f16 ----------
// V is written TRANSPOSED per batch: vt[batch][ch][n], so the attention kernel
// can DMA contiguous [ch][key-chunk] tiles straight into LDS.
__global__ void __launch_bounds__(64)
proj_qkv(const float* __restrict__ x,
         const float* __restrict__ Wf, const float* __restrict__ bf,
         const float* __restrict__ Wg, const float* __restrict__ bg,
         const float* __restrict__ Wh, const float* __restrict__ bh,
         h16* __restrict__ qp, h16* __restrict__ kp, h16* __restrict__ vt) {
    const int row = blockIdx.x;      // 0 .. B*N-1
    const int c   = threadIdx.x;     // 0 .. 63
    __shared__ float xs[CDIM];
    xs[c] = x[row * CDIM + c];
    __syncthreads();

    float acc = bh[c];
#pragma unroll
    for (int i = 0; i < CDIM; ++i) acc = fmaf(xs[i], Wh[i * CDIM + c], acc);
    const int batch = row >> 12;                 // / NSEQ
    const int n     = row & (NSEQ - 1);
    vt[((size_t)batch * CDIM + c) * NSEQ + n] = (h16)acc;

    if (c < DPAD) {
        float q = 0.f, k = 0.f;
        if (c < DHEAD) {
            q = bf[c]; k = bg[c];
#pragma unroll
            for (int i = 0; i < CDIM; ++i) {
                q = fmaf(xs[i], Wf[i * DHEAD + c], q);
                k = fmaf(xs[i], Wg[i * DHEAD + c], k);
            }
        }
        qp[row * DPAD + c] = (h16)q;   // zeros for c in [8,32)
        kp[row * DPAD + c] = (h16)k;
    }
}

// ---------------- kernel 2: flash attention, TDM double-buffered ---------------
__global__ void __launch_bounds__(128)
flash_attn(const float* __restrict__ x, const float* __restrict__ gamma_p,
           const h16* __restrict__ qp, const h16* __restrict__ kp,
           const h16* __restrict__ vt, float* __restrict__ out) {
    const int tid  = threadIdx.x;
    const int lane = tid & 31;
    const int wave = tid >> 5;
    const int half = lane >> 4;        // C-layout: lanes 0-15 hold M=r, 16-31 hold M=r+8
    const int l16  = lane & 15;
    const int batch = blockIdx.x >> 6;                         // 64 blocks per batch
    const int qbase = batch * NSEQ + (blockIdx.x & 63) * BM + wave * 16;
    const int kbase = batch * NSEQ;

    __shared__ __align__(64) h16 k_lds[2][BK * DPAD];          // [buf][key][dpad]   2x4KB
    __shared__ __align__(64) h16 v_lds[2][CDIM * BK];          // [buf][ch][key]     2x8KB
    __shared__ __align__(64) h16 p_lds[NWAVE * 16 * BK];       // per-wave P [row][key] 8KB
    h16* pw = p_lds + wave * (16 * BK);

    // A operand: 16x32 Q tile (head dim zero-padded to 32)
    const v16h aQ = *reinterpret_cast<const v16h*>(qp + (qbase + l16) * DPAD + half * 16);

    v8f acc0 = {}, acc1 = {}, acc2 = {}, acc3 = {};            // O tile 16 x 64 (f32)
    float m[8], l[8];
#pragma unroll
    for (int r = 0; r < 8; ++r) { m[r] = -1e30f; l[r] = 0.f; }

#if HAVE_TDM
    if (wave == 0) {  // prologue: DMA chunk 0 into buffer 0 (TDM ignores EXEC)
        tdm_load_2d((unsigned)(size_t)(const void*)k_lds[0],
                    kp + (size_t)kbase * DPAD, BK * DPAD, 1, BK * DPAD, 1, BK * DPAD);
        tdm_load_2d((unsigned)(size_t)(const void*)v_lds[0],
                    vt + (size_t)batch * CDIM * NSEQ, BK, CDIM, NSEQ, CDIM, NSEQ);
    }
#else
    {   // prologue fallback: cooperative copy of chunk 0 into buffer 0
        const uint4* ks = reinterpret_cast<const uint4*>(kp + (size_t)kbase * DPAD);
        uint4* kd = reinterpret_cast<uint4*>(k_lds[0]);
        kd[tid] = ks[tid]; kd[tid + 128] = ks[tid + 128];
        uint4* vd = reinterpret_cast<uint4*>(v_lds[0]);
#pragma unroll
        for (int i = 0; i < 4; ++i) {
            const int e = tid + i * 128, ch = e >> 3, part = e & 7;
            vd[e] = *reinterpret_cast<const uint4*>(
                vt + ((size_t)batch * CDIM + ch) * NSEQ + part * 8);
        }
    }
#endif

    for (int j = 0; j < NSEQ; j += BK) {
        const int b = (j / BK) & 1;
#if HAVE_TDM
        if (wave == 0) __builtin_amdgcn_s_wait_tensorcnt(0);   // chunk j landed in LDS
#endif
        __syncthreads();

        if (j + BK < NSEQ) {   // stage chunk j+1 into the other buffer while computing
#if HAVE_TDM
            if (wave == 0) {
                tdm_load_2d((unsigned)(size_t)(const void*)k_lds[b ^ 1],
                            kp + (size_t)(kbase + j + BK) * DPAD, BK * DPAD, 1,
                            BK * DPAD, 1, BK * DPAD);
                tdm_load_2d((unsigned)(size_t)(const void*)v_lds[b ^ 1],
                            vt + (size_t)batch * CDIM * NSEQ + (j + BK), BK, CDIM,
                            NSEQ, CDIM, NSEQ);
            }
#else
            const uint4* ks = reinterpret_cast<const uint4*>(kp + (size_t)(kbase + j + BK) * DPAD);
            uint4* kd = reinterpret_cast<uint4*>(k_lds[b ^ 1]);
            kd[tid] = ks[tid]; kd[tid + 128] = ks[tid + 128];
            uint4* vd = reinterpret_cast<uint4*>(v_lds[b ^ 1]);
#pragma unroll
            for (int i = 0; i < 4; ++i) {
                const int e = tid + i * 128, ch = e >> 3, part = e & 7;
                vd[e] = *reinterpret_cast<const uint4*>(
                    vt + ((size_t)batch * CDIM + ch) * NSEQ + (j + BK) + part * 8);
            }
#endif
        }

        // ---- S = Q * K^T : four 16-key tiles ----
        v8f s[4];
#pragma unroll
        for (int t = 0; t < 4; ++t) {
            const v16h bK = *reinterpret_cast<const v16h*>(
                k_lds[b] + (t * 16 + l16) * DPAD + half * 16);
            v8f z = {};
            s[t] = __builtin_amdgcn_wmma_f32_16x16x32_f16(false, aQ, false, bK,
                                                          (short)0, z, false, false);
        }

        // ---- online softmax over 64 keys (rows r / r+8 live in lane halves) ----
#pragma unroll
        for (int r = 0; r < 8; ++r) {
            const float tm = rmax16(fmaxf(fmaxf(s[0][r], s[1][r]), fmaxf(s[2][r], s[3][r])));
            const float mn = fmaxf(m[r], tm);
            const float sc = __expf(m[r] - mn);
            m[r] = mn;
            float rs = 0.f;
#pragma unroll
            for (int t = 0; t < 4; ++t) {
                const float p = __expf(s[t][r] - mn);
                s[t][r] = p;                       // reuse s[] to hold P
                rs += p;
            }
            l[r] = l[r] * sc + rsum16(rs);
            acc0[r] *= sc; acc1[r] *= sc; acc2[r] *= sc; acc3[r] *= sc;
        }

        // ---- stage P as f16 through per-wave LDS (C-layout -> A-layout) ----
#pragma unroll
        for (int r = 0; r < 8; ++r) {
            const int prow = (r + half * 8) * BK;
#pragma unroll
            for (int t = 0; t < 4; ++t) pw[prow + t * 16 + l16] = (h16)s[t][r];
        }
#if __has_builtin(__builtin_amdgcn_s_wait_dscnt)
        __builtin_amdgcn_s_wait_dscnt(0);          // intra-wave DS store->load ordering
#else
        __threadfence_block();
#endif
        const v16h aP0 = *reinterpret_cast<const v16h*>(pw + l16 * BK + half * 16);
        const v16h aP1 = *reinterpret_cast<const v16h*>(pw + l16 * BK + 32 + half * 16);

        // ---- O += P * V : 4 channel tiles x 2 key halves ----
#pragma unroll
        for (int ct = 0; ct < 4; ++ct) {
            const v16h bV0 = *reinterpret_cast<const v16h*>(
                v_lds[b] + (ct * 16 + l16) * BK + half * 16);
            const v16h bV1 = *reinterpret_cast<const v16h*>(
                v_lds[b] + (ct * 16 + l16) * BK + 32 + half * 16);
            v8f* a = (ct == 0) ? &acc0 : (ct == 1) ? &acc1 : (ct == 2) ? &acc2 : &acc3;
            *a = __builtin_amdgcn_wmma_f32_16x16x32_f16(false, aP0, false, bV0,
                                                        (short)0, *a, false, false);
            *a = __builtin_amdgcn_wmma_f32_16x16x32_f16(false, aP1, false, bV1,
                                                        (short)0, *a, false, false);
        }
    }

    // ---- epilogue: out = gamma * (O / l) + x ----
    const float g = gamma_p[0];
#pragma unroll
    for (int r = 0; r < 8; ++r) {
        const int row = qbase + r + half * 8;
        const float inv = g / l[r];
        out[row * CDIM + ( 0 + l16)] = acc0[r] * inv + x[row * CDIM + ( 0 + l16)];
        out[row * CDIM + (16 + l16)] = acc1[r] * inv + x[row * CDIM + (16 + l16)];
        out[row * CDIM + (32 + l16)] = acc2[r] * inv + x[row * CDIM + (32 + l16)];
        out[row * CDIM + (48 + l16)] = acc3[r] * inv + x[row * CDIM + (48 + l16)];
    }
}

extern "C" void kernel_launch(void* const* d_in, const int* in_sizes, int n_in,
                              void* d_out, int out_size, void* d_ws, size_t ws_size,
                              hipStream_t stream) {
    const float* x     = (const float*)d_in[0];
    const float* Wf    = (const float*)d_in[1];
    const float* bf    = (const float*)d_in[2];
    const float* Wg    = (const float*)d_in[3];
    const float* bg    = (const float*)d_in[4];
    const float* Wh    = (const float*)d_in[5];
    const float* bh    = (const float*)d_in[6];
    const float* gamma = (const float*)d_in[7];
    float* out = (float*)d_out;

    h16* qp = (h16*)d_ws;                                   // [B*N][32] padded Q (1 MB)
    h16* kp = qp + (size_t)BATCH * NSEQ * DPAD;             // [B*N][32] padded K (1 MB)
    h16* vt = kp + (size_t)BATCH * NSEQ * DPAD;             // [B][64][N] V^T     (2 MB)

    proj_qkv<<<BATCH * NSEQ, 64, 0, stream>>>(x, Wf, bf, Wg, bg, Wh, bh, qp, kp, vt);
    flash_attn<<<BATCH * (NSEQ / BM), 128, 0, stream>>>(x, gamma, qp, kp, vt, out);
}
